// Model_66048007078099
// MI455X (gfx1250) — compile-verified, matrix operand order checked by reading, not run on hardware
//
#include <hip/hip_runtime.h>
#include <hip/hip_bf16.h>
#include <stdint.h>

// ---------------------------------------------------------------------------
// MI455X (gfx1250): bf16 WMMA GEMMs (2x4 register-blocked), persistent
// grid-synced GRU recurrence with TDM (tensor_load_to_lds) weight staging.
// ---------------------------------------------------------------------------

#define BN_INV 0.99999500003749969f  // 1/sqrt(1 + 1e-5)

typedef float  v8f   __attribute__((ext_vector_type(8)));
typedef __bf16 v16bf __attribute__((ext_vector_type(16)));
typedef unsigned u32x4 __attribute__((ext_vector_type(4)));
typedef int      i32x8 __attribute__((ext_vector_type(8)));
typedef int      i32x4 __attribute__((ext_vector_type(4)));

union FragBF { uint4 q[2]; v16bf v; };

// A-fragment (16x32 bf16, row-major) per ISA layout:
//   lane = m + 16*khalf ; VGPR0..3 <- K in [8*khalf, 8*khalf+8)
//                         VGPR4..7 <- K in [16+8*khalf, 16+8*khalf+8)
// B-fragment (32x16, B[k][n] = W[n][k]) loads identically from row-major W[N,K].
__device__ __forceinline__ v16bf load_frag_bf16(const __bf16* __restrict__ base,
                                                int row0, long ld, int col0, int lane) {
  int r  = row0 + (lane & 15);
  int kh = (lane >> 4) << 3;
  const __bf16* p = base + (long)r * ld + col0 + kh;
  FragBF f;
  f.q[0] = *(const uint4*)(p);
  f.q[1] = *(const uint4*)(p + 16);
  return f.v;
}

__device__ __forceinline__ v8f wmma_bf16(v16bf a, v16bf b, v8f c) {
  return __builtin_amdgcn_wmma_f32_16x16x32_bf16(false, a, false, b, (short)0, c,
                                                 false, false);
}

__device__ __forceinline__ float sigmoidf_(float x) { return 1.f / (1.f + __expf(-x)); }

// ---------------------------------------------------------------------------
// GEMM: Y[M,N] = act(A_bf16[M,K] @ W_bf16[N,K]^T + bias)
// 2x4 register blocking: each wave owns a 32x64 output block (8 wmma/k-step,
// 6 fragment loads/k-step). Mp = M/32, Nq = N/64. 8 waves per 256-thr block.
// ACT: 0=none 1=relu.  BIAS: 0=none 1=per-column.
// ---------------------------------------------------------------------------
template <int ACT, int BIAS>
__global__ __launch_bounds__(256) void gemm_bf16_k(
    const __bf16* __restrict__ A, long ldA,
    const __bf16* __restrict__ Bw, long ldB,
    const float* __restrict__ bias,
    float* __restrict__ Y, long ldY,
    int Mp, int Nq, int K) {
  int wave = blockIdx.x * 8 + (threadIdx.x >> 5);
  if (wave >= Mp * Nq) return;  // whole-wave uniform exit
  int lane = threadIdx.x & 31;
  int mp = wave / Nq, nq = wave - mp * Nq;
  int m0 = mp * 32, n0 = nq * 64;
  v8f acc[2][4] = {};
  for (int k = 0; k < K; k += 32) {
    v16bf a0 = load_frag_bf16(A, m0,      ldA, k, lane);
    v16bf a1 = load_frag_bf16(A, m0 + 16, ldA, k, lane);
#pragma unroll
    for (int j = 0; j < 4; ++j) {
      v16bf b = load_frag_bf16(Bw, n0 + 16 * j, ldB, k, lane);
      acc[0][j] = wmma_bf16(a0, b, acc[0][j]);
      acc[1][j] = wmma_bf16(a1, b, acc[1][j]);
    }
  }
  int nl = lane & 15;
  int mrow = (lane >> 4) << 3;
#pragma unroll
  for (int i = 0; i < 2; ++i) {
#pragma unroll
    for (int j = 0; j < 4; ++j) {
      int n = n0 + 16 * j + nl;
      int mb = m0 + 16 * i + mrow;
      float bc = (BIAS == 1) ? bias[n] : 0.f;
#pragma unroll
      for (int r = 0; r < 8; ++r) {
        float v = acc[i][j][r] + bc;
        if (ACT == 1) v = fmaxf(v, 0.f);
        Y[(long)(mb + r) * ldY + n] = v;
      }
    }
  }
}

// ---------------------------------------------------------------------------
// Batched GEMM for fl linear: per batch b,
//   cat[b, :, 512:1024] = fl_W[480,128p] @ yfT[b][512,128p]^T + fl_b (per-row)
// ---------------------------------------------------------------------------
__global__ __launch_bounds__(256) void gemm_fl_k(
    const __bf16* __restrict__ A,       // fl_W bf16 [480,128] (K padded)
    const __bf16* __restrict__ Bw,      // yfT bf16 [64,512,128]
    const float* __restrict__ biasRow,  // fl_b [480]
    float* __restrict__ cat) {
  const int Mp = 15, Nq = 8;  // 480/32, 512/64
  int wave = blockIdx.x * 8 + (threadIdx.x >> 5);
  if (wave >= 64 * Mp * Nq) return;
  int lane = threadIdx.x & 31;
  int bidx = wave / (Mp * Nq);
  int rem  = wave - bidx * (Mp * Nq);
  int mp = rem / Nq, nq = rem - (rem / Nq) * Nq;
  int m0 = mp * 32, n0 = nq * 64;
  const __bf16* Bb = Bw + (size_t)bidx * 512 * 128;
  v8f acc[2][4] = {};
  for (int k = 0; k < 128; k += 32) {
    v16bf a0 = load_frag_bf16(A, m0,      128, k, lane);
    v16bf a1 = load_frag_bf16(A, m0 + 16, 128, k, lane);
#pragma unroll
    for (int j = 0; j < 4; ++j) {
      v16bf b = load_frag_bf16(Bb, n0 + 16 * j, 128, k, lane);
      acc[0][j] = wmma_bf16(a0, b, acc[0][j]);
      acc[1][j] = wmma_bf16(a1, b, acc[1][j]);
    }
  }
  int nl = lane & 15;
  int mrow = (lane >> 4) << 3;
  float* Y = cat + (size_t)bidx * 480 * 1024 + 512;
#pragma unroll
  for (int i = 0; i < 2; ++i) {
#pragma unroll
    for (int j = 0; j < 4; ++j) {
      int n = n0 + 16 * j + nl;
      int mb = m0 + 16 * i + mrow;
#pragma unroll
      for (int r = 0; r < 8; ++r)
        Y[(long)(mb + r) * 1024 + n] = acc[i][j][r] + biasRow[mb + r];
    }
  }
}

// ---------------------------------------------------------------------------
// Persistent GRU recurrence. Grid = 16 WGs x 256 thr; WG g owns hidden cols
// [32g, 32g+32). At startup the WG's Whh slice (3 gates x 32 rows x 512 K,
// 96 KB bf16) is staged into LDS with 3 TDM descriptors (tensor_load_to_lds)
// and reused for all T steps. Per step each wave computes one (batch-tile,
// hidden-tile) pair = 3 gate tiles, gate math, h_t write, grid barrier.
// Dynamic LDS: [96][512] bf16 Whh | [64][32] f32 hprev | [3][32] f32 bhh.
// ---------------------------------------------------------------------------
__global__ __launch_bounds__(256) void gru_rec_k(
    const float* __restrict__ gi,    // [64, T, 1536]
    const __bf16* __restrict__ Whh,  // [1536, 512] bf16
    const float* __restrict__ bhh,   // [1536]
    __bf16* __restrict__ ybf,        // [64, T, 512] bf16
    float* __restrict__ yf32,        // optional f32 copy
    int T, unsigned* __restrict__ ctr) {
  const int H = 512;
  extern __shared__ char smem[];
  __bf16* whh_s = (__bf16*)smem;                    // [96][512]
  float*  hprev = (float*)(smem + 96 * 512 * 2);    // [64][32]
  float*  bhs   = hprev + 64 * 32;                  // [3][32]
  int tid = threadIdx.x, lane = tid & 31, wv = tid >> 5;
  int hbase = blockIdx.x * 32;

  // --- TDM: stage Whh slice into LDS (one descriptor per gate) ---
  if (wv == 0) {
#pragma unroll
    for (int g = 0; g < 3; ++g) {
      unsigned long long ga =
          (unsigned long long)(uintptr_t)(Whh + ((size_t)g * H + hbase) * H);
      u32x4 d0;
      d0[0] = 1u;                                   // count=1, load descriptor
      d0[1] = (unsigned)(g * 32 * H * 2);           // lds_addr (bytes)
      d0[2] = (unsigned)(ga & 0xFFFFFFFFu);         // global_addr[31:0]
      d0[3] = (unsigned)((ga >> 32) & 0x01FFFFFFu) | (2u << 30);  // addr[56:32]|type=2
      i32x8 d1 = {};
      d1[0] = 0x00010000;                           // data_size = 2 bytes
      d1[1] = (int)(512u << 16);                    // tensor_dim0 = 512 (lo16)
      d1[2] = (int)(32u << 16);                     // tensor_dim1 = 32  (lo16)
      d1[3] = (int)(512u << 16);                    // tile_dim0 = 512
      d1[4] = 32;                                   // tile_dim1 = 32
      d1[5] = 512;                                  // tensor_dim0_stride = 512
      i32x4 z4 = {};
      i32x8 z8 = {};
      __builtin_amdgcn_tensor_load_to_lds(d0, d1, z4, z4, z8, 0);
    }
    __builtin_amdgcn_s_wait_tensorcnt(0);
  }
  for (int i = tid; i < 64 * 32; i += 256) hprev[i] = 0.f;
  for (int i = tid; i < 96; i += 256)
    bhs[i] = bhh[(i >> 5) * H + hbase + (i & 31)];
  __syncthreads();

  int mt  = wv & 3;        // batch tile (M=64 -> 4 tiles)
  int htl = wv >> 2;       // local hidden tile (2 per WG)
  for (int t = 0; t < T; ++t) {
    v8f ar = {}, az = {}, an = {};
    if (t > 0) {
      const __bf16* hb = ybf + (long)(t - 1) * H;  // row b at stride T*H
      for (int k = 0; k < H; k += 32) {
        v16bf a  = load_frag_bf16(hb, mt * 16, (long)T * H, k, lane);
        v16bf br = load_frag_bf16(whh_s, 0 * 32 + htl * 16, H, k, lane);
        v16bf bz = load_frag_bf16(whh_s, 1 * 32 + htl * 16, H, k, lane);
        v16bf bn = load_frag_bf16(whh_s, 2 * 32 + htl * 16, H, k, lane);
        ar = wmma_bf16(a, br, ar);
        az = wmma_bf16(a, bz, az);
        an = wmma_bf16(a, bn, an);
      }
    }
    int nl = lane & 15;
    int mrow = (lane >> 4) << 3;
    int jl = htl * 16 + nl;   // local hidden col (0..31)
    int j  = hbase + jl;
#pragma unroll
    for (int r = 0; r < 8; ++r) {
      int bb = mt * 16 + mrow + r;
      long gir = ((long)bb * T + t) * (3 * H);
      float ghr = ar[r] + bhs[0 * 32 + jl];
      float ghz = az[r] + bhs[1 * 32 + jl];
      float ghn = an[r] + bhs[2 * 32 + jl];
      float rr = sigmoidf_(gi[gir + j] + ghr);
      float zz = sigmoidf_(gi[gir + H + j] + ghz);
      float nn = tanhf(gi[gir + 2 * H + j] + rr * ghn);
      float hp = hprev[bb * 32 + jl];
      float hn = (1.f - zz) * nn + zz * hp;
      hprev[bb * 32 + jl] = hn;
      long yi = ((long)bb * T + t) * H + j;
      ybf[yi] = (__bf16)hn;
      if (yf32) yf32[yi] = hn;
    }
    // grid barrier (monotonic counter, 16 co-resident WGs)
    __syncthreads();
    if (tid == 0) {
      __threadfence();
      atomicAdd(ctr, 1u);
      unsigned tgt = (unsigned)(t + 1) * gridDim.x;
      while (__hip_atomic_load(ctr, __ATOMIC_ACQUIRE, __HIP_MEMORY_SCOPE_AGENT) < tgt)
        __builtin_amdgcn_s_sleep(1);
    }
    __syncthreads();
  }
}

// ------------------------------ elementwise --------------------------------
__global__ void cvt_bf16_k(const float* __restrict__ s, __bf16* __restrict__ d, long n) {
  long i = (long)blockIdx.x * 256 + threadIdx.x;
  if (i < n) d[i] = (__bf16)s[i];
}

__global__ void cvt_fl_k(const float* __restrict__ s, __bf16* __restrict__ d) {
  int i = blockIdx.x * 256 + threadIdx.x;  // 480*128, K padded 120->128
  if (i >= 480 * 128) return;
  int r = i >> 7, c = i & 127;
  d[i] = (c < 120) ? (__bf16)s[r * 120 + c] : (__bf16)0.f;
}

__global__ void bn_cvt_k(const float* __restrict__ s, const float* __restrict__ g,
                         const float* __restrict__ bb, __bf16* __restrict__ d,
                         long n, int Cmask) {
  long i = (long)blockIdx.x * 256 + threadIdx.x;
  if (i >= n) return;
  int c = (int)(i & Cmask);
  d[i] = (__bf16)(s[i] * (g[c] * BN_INV) + bb[c]);
}

__global__ void t_to_cat_k(const float* __restrict__ y, const float* __restrict__ g,
                           const float* __restrict__ bb, float* __restrict__ cat) {
  long i = (long)blockIdx.x * 256 + threadIdx.x;  // 30720*512
  if (i >= (long)30720 * 512) return;
  long m = i >> 9; int c = (int)(i & 511);
  cat[m * 1024 + c] = y[i] * (g[c] * BN_INV) + bb[c];
}

__global__ void f_to_yfT_k(const float* __restrict__ y, const float* __restrict__ g,
                           const float* __restrict__ bb, __bf16* __restrict__ d) {
  long i = (long)blockIdx.x * 256 + threadIdx.x;  // 64*512*128
  if (i >= (long)64 * 512 * 128) return;
  int k = (int)(i & 127);
  long t2 = i >> 7;
  int h = (int)(t2 & 511);
  int bi = (int)(t2 >> 9);
  float v = 0.f;
  if (k < 120) v = y[((long)bi * 120 + k) * 512 + h] * (g[h] * BN_INV) + bb[h];
  d[i] = (__bf16)v;
}

// ------------------------------- decoder -----------------------------------
__global__ __launch_bounds__(256) void dec_px_k(
    const float* __restrict__ h2, const float* __restrict__ g,
    const float* __restrict__ bb, const float* __restrict__ pW,
    const float* __restrict__ pb, float* __restrict__ px, int M) {
  int row = blockIdx.x * 8 + (threadIdx.x >> 5);
  if (row >= M) return;
  int lane = threadIdx.x & 31;
  float s = 0.f;
#pragma unroll
  for (int c = lane; c < 512; c += 32) {
    float x = h2[(long)row * 512 + c] * (g[c] * BN_INV) + bb[c];
    s += x * pW[c];
  }
#pragma unroll
  for (int o = 16; o > 0; o >>= 1) s += __shfl_xor(s, o, 32);
  if (lane == 0) px[row] = s + pb[0];
}

__global__ __launch_bounds__(256) void dec_rec_k(
    const float* __restrict__ px, const float* __restrict__ Wih,
    const float* __restrict__ bih, const float* __restrict__ bhh,
    const float* __restrict__ pW, const float* __restrict__ pb,
    float* __restrict__ out, int T) {
  int wv = threadIdx.x >> 5, lane = threadIdx.x & 31;
  int b = blockIdx.x * 8 + wv;
  float Wr[16], Wz[16], Wn[16], Br[16], Bz[16], Bn[16], Hn[16], Pw[16];
#pragma unroll
  for (int e = 0; e < 16; ++e) {
    int j = lane + 32 * e;
    Wr[e] = Wih[j]; Wz[e] = Wih[512 + j]; Wn[e] = Wih[1024 + j];
    Br[e] = bih[j] + bhh[j];
    Bz[e] = bih[512 + j] + bhh[512 + j];
    Bn[e] = bih[1024 + j];
    Hn[e] = bhh[1024 + j];
    Pw[e] = pW[j];
  }
  float f = 0.f;
  float pbv = pb[0];
  for (int t = 0; t < T; ++t) {
    float inp = (f + px[b * T + t]) * 0.5f;
    float s = 0.f;
#pragma unroll
    for (int e = 0; e < 16; ++e) {
      float r = sigmoidf_(inp * Wr[e] + Br[e]);
      float z = sigmoidf_(inp * Wz[e] + Bz[e]);
      float n = tanhf(inp * Wn[e] + Bn[e] + r * Hn[e]);
      s += (1.f - z) * n * Pw[e];
    }
#pragma unroll
    for (int o = 16; o > 0; o >>= 1) s += __shfl_xor(s, o, 32);
    f = s + pbv;
    if (lane == 0) out[b * T + t] = f;
  }
}

// ---------------------------------------------------------------------------
extern "C" void kernel_launch(void* const* d_in, const int* in_sizes, int n_in,
                              void* d_out, int out_size, void* d_ws, size_t ws_size,
                              hipStream_t stream) {
  (void)in_sizes; (void)n_in; (void)out_size; (void)ws_size;
  const float* x_time  = (const float*)d_in[0];
  const float* x_freq  = (const float*)d_in[1];
  const float* t_Wih0  = (const float*)d_in[2];
  const float* t_Whh0  = (const float*)d_in[3];
  const float* t_bih0  = (const float*)d_in[4];
  const float* t_bhh0  = (const float*)d_in[5];
  const float* t_Wih1  = (const float*)d_in[6];
  const float* t_Whh1  = (const float*)d_in[7];
  const float* t_bih1  = (const float*)d_in[8];
  const float* t_bhh1  = (const float*)d_in[9];
  const float* t_bn_g  = (const float*)d_in[10];
  const float* t_bn_b  = (const float*)d_in[11];
  const float* f_Wih0  = (const float*)d_in[12];
  const float* f_Whh0  = (const float*)d_in[13];
  const float* f_bih0  = (const float*)d_in[14];
  const float* f_bhh0  = (const float*)d_in[15];
  const float* f_Wih1  = (const float*)d_in[16];
  const float* f_Whh1  = (const float*)d_in[17];
  const float* f_bih1  = (const float*)d_in[18];
  const float* f_bhh1  = (const float*)d_in[19];
  const float* f_bn_g  = (const float*)d_in[20];
  const float* f_bn_b  = (const float*)d_in[21];
  const float* fl_W    = (const float*)d_in[22];
  const float* fl_b    = (const float*)d_in[23];
  const float* ag_bn1_g = (const float*)d_in[24];
  const float* ag_bn1_b = (const float*)d_in[25];
  const float* ag_W1   = (const float*)d_in[26];
  const float* ag_b1   = (const float*)d_in[27];
  const float* ag_bn2_g = (const float*)d_in[28];
  const float* ag_bn2_b = (const float*)d_in[29];
  const float* ag_W2   = (const float*)d_in[30];
  const float* ag_b2   = (const float*)d_in[31];
  const float* dec_bn_g = (const float*)d_in[32];
  const float* dec_bn_b = (const float*)d_in[33];
  const float* dec_Wih = (const float*)d_in[34];
  const float* dec_bih = (const float*)d_in[35];
  const float* dec_bhh = (const float*)d_in[36];
  const float* dec_pW  = (const float*)d_in[37];
  const float* dec_pb  = (const float*)d_in[38];

  // --- workspace layout -----------------------------------------------------
  char* ws = (char*)d_ws;
  size_t o = 0;
  auto alloc = [&](size_t bytes) {
    o = (o + 255) & ~(size_t)255; size_t r = o; o += bytes; return r;
  };
  auto bfp  = [&](size_t e) { return (__bf16*)(ws + alloc(e * 2)); };
  auto f32p = [&](size_t e) { return (float*)(ws + alloc(e * 4)); };

  __bf16* tWih0b = bfp((size_t)1536 * 128);
  __bf16* tWhh0b = bfp((size_t)1536 * 512);
  __bf16* tWih1b = bfp((size_t)1536 * 512);
  __bf16* tWhh1b = bfp((size_t)1536 * 512);
  __bf16* fWih0b = bfp((size_t)1536 * 128);
  __bf16* fWhh0b = bfp((size_t)1536 * 512);
  __bf16* fWih1b = bfp((size_t)1536 * 512);
  __bf16* fWhh1b = bfp((size_t)1536 * 512);
  __bf16* flWb   = bfp((size_t)480 * 128);
  __bf16* agW1b  = bfp((size_t)3 * 512 * 1024);
  __bf16* agW2b  = bfp((size_t)3 * 512 * 512);
  __bf16* xtb    = bfp((size_t)30720 * 128);
  __bf16* xfb    = bfp((size_t)7680 * 128);
  float*  gi     = f32p((size_t)30720 * 1536);  // reused as `cat` later
  __bf16* y0tb   = bfp((size_t)30720 * 512);
  __bf16* y1tb   = bfp((size_t)30720 * 512);
  float*  y1tf   = f32p((size_t)30720 * 512);
  __bf16* y0fb   = bfp((size_t)7680 * 512);
  __bf16* y1fb   = bfp((size_t)7680 * 512);
  float*  y1ff   = f32p((size_t)7680 * 512);
  __bf16* yfTb   = bfp((size_t)64 * 512 * 128);
  __bf16* catbnb = bfp((size_t)30720 * 1024);
  float*  h1f    = f32p((size_t)30720 * 512);
  __bf16* h1bnb  = bfp((size_t)30720 * 512);
  float*  h2f    = f32p((size_t)30720 * 512);
  float*  pxf    = f32p(30720);
  unsigned* ctr  = (unsigned*)(ws + alloc(256));
  float* cat = gi;

  (void)hipMemsetAsync(ctr, 0, 256, stream);

  auto cvt = [&](const float* s, __bf16* d, long n) {
    cvt_bf16_k<<<(unsigned)((n + 255) / 256), 256, 0, stream>>>(s, d, n);
  };
  cvt(t_Wih0, tWih0b, 1536L * 128); cvt(t_Whh0, tWhh0b, 1536L * 512);
  cvt(t_Wih1, tWih1b, 1536L * 512); cvt(t_Whh1, tWhh1b, 1536L * 512);
  cvt(f_Wih0, fWih0b, 1536L * 128); cvt(f_Whh0, fWhh0b, 1536L * 512);
  cvt(f_Wih1, fWih1b, 1536L * 512); cvt(f_Whh1, fWhh1b, 1536L * 512);
  cvt(ag_W1, agW1b, 3L * 512 * 1024); cvt(ag_W2, agW2b, 3L * 512 * 512);
  cvt(x_time, xtb, 30720L * 128); cvt(x_freq, xfb, 7680L * 128);
  cvt_fl_k<<<(480 * 128 + 255) / 256, 256, 0, stream>>>(fl_W, flWb);

  const unsigned GRU_LDS = 96 * 512 * 2 + 64 * 32 * 4 + 96 * 4;  // 106880 B

  // --- time branch ---
  { int waves = 960 * 24;
    gemm_bf16_k<0, 1><<<(waves + 7) / 8, 256, 0, stream>>>(
        xtb, 128, tWih0b, 128, t_bih0, gi, 1536, 960, 24, 128); }
  gru_rec_k<<<16, 256, GRU_LDS, stream>>>(gi, tWhh0b, t_bhh0, y0tb, nullptr, 480, ctr + 0);
  { int waves = 960 * 24;
    gemm_bf16_k<0, 1><<<(waves + 7) / 8, 256, 0, stream>>>(
        y0tb, 512, tWih1b, 512, t_bih1, gi, 1536, 960, 24, 512); }
  gru_rec_k<<<16, 256, GRU_LDS, stream>>>(gi, tWhh1b, t_bhh1, y1tb, y1tf, 480, ctr + 1);

  // --- freq branch ---
  { int waves = 240 * 24;
    gemm_bf16_k<0, 1><<<(waves + 7) / 8, 256, 0, stream>>>(
        xfb, 128, fWih0b, 128, f_bih0, gi, 1536, 240, 24, 128); }
  gru_rec_k<<<16, 256, GRU_LDS, stream>>>(gi, fWhh0b, f_bhh0, y0fb, nullptr, 120, ctr + 2);
  { int waves = 240 * 24;
    gemm_bf16_k<0, 1><<<(waves + 7) / 8, 256, 0, stream>>>(
        y0fb, 512, fWih1b, 512, f_bih1, gi, 1536, 240, 24, 512); }
  gru_rec_k<<<16, 256, GRU_LDS, stream>>>(gi, fWhh1b, f_bhh1, y1fb, y1ff, 120, ctr + 3);

  // --- cat = [bn(t) | fl(bn(f))] ---  (gi buffer reused as cat from here)
  t_to_cat_k<<<(unsigned)((30720L * 512 + 255) / 256), 256, 0, stream>>>(
      y1tf, t_bn_g, t_bn_b, cat);
  f_to_yfT_k<<<(unsigned)((64L * 512 * 128 + 255) / 256), 256, 0, stream>>>(
      y1ff, f_bn_g, f_bn_b, yfTb);
  { int waves = 64 * 15 * 8;
    gemm_fl_k<<<(waves + 7) / 8, 256, 0, stream>>>(flWb, yfTb, fl_b, cat); }

  // --- aggregators + decoders (per source) ---
  for (int s = 0; s < 3; ++s) {
    bn_cvt_k<<<(unsigned)((30720L * 1024 + 255) / 256), 256, 0, stream>>>(
        cat, ag_bn1_g + s * 1024, ag_bn1_b + s * 1024, catbnb, 30720L * 1024, 1023);
    { int waves = 960 * 8;
      gemm_bf16_k<1, 1><<<(waves + 7) / 8, 256, 0, stream>>>(
          catbnb, 1024, agW1b + (size_t)s * 512 * 1024, 1024,
          ag_b1 + s * 512, h1f, 512, 960, 8, 1024); }
    bn_cvt_k<<<(unsigned)((30720L * 512 + 255) / 256), 256, 0, stream>>>(
        h1f, ag_bn2_g + s * 512, ag_bn2_b + s * 512, h1bnb, 30720L * 512, 511);
    { int waves = 960 * 8;
      gemm_bf16_k<1, 1><<<(waves + 7) / 8, 256, 0, stream>>>(
          h1bnb, 512, agW2b + (size_t)s * 512 * 512, 512,
          ag_b2 + s * 512, h2f, 512, 960, 8, 512); }
    dec_px_k<<<30720 / 8, 256, 0, stream>>>(
        h2f, dec_bn_g + s * 512, dec_bn_b + s * 512, dec_pW + s * 512,
        dec_pb + s, pxf, 30720);
    dec_rec_k<<<8, 256, 0, stream>>>(
        pxf, dec_Wih + s * 1536, dec_bih + s * 1536, dec_bhh + s * 1536,
        dec_pW + s * 512, dec_pb + s, (float*)d_out + (size_t)s * 30720, 480);
  }
}
